// BitLinear_83803401879884
// MI455X (gfx1250) — compile-verified, matrix operand order checked by reading, not run on hardware
//
#include <hip/hip_runtime.h>

// BitLinear (ternary weight x int8 activation) for MI455X / gfx1250.
// Exact integer core: y = (x_int8 . w_ternary) * (x_scale * w_scale)
// computed with V_WMMA_I32_16X16X64_IU8 (wave32).
//
// Quantized operands are stored FRAGMENT-MAJOR: for each 16x64 int8 tile
// (frag), lane L's 32 bytes -- K chunks {0-7,16-23,32-39,48-55} for L<16,
// {8-15,24-31,40-47,56-63} for L>=16, exactly the ISA 8-bit A/B VGPR
// layout -- are contiguous at frag*1024 + L*32. The GEMM then loads each
// fragment with two global_load_b128 (vs four b64 for row-major).

typedef int v8i __attribute__((ext_vector_type(8)));

static constexpr int M_DIM = 8192;    // 4 * 2048
static constexpr int K_DIM = 4096;
static constexpr int N_DIM = 11008;
static constexpr int KT = K_DIM / 64;                  // 64 k-tiles
static constexpr size_t FRAG_BYTES = 1024;             // one 16x64 int8 tile
static constexpr size_t MT_STRIDE = (size_t)KT * FRAG_BYTES;  // 65536
static constexpr int RED_BLOCKS = 1024;
static constexpr float EPSQ = 1e-5f;

// ---------------- workspace layout (bytes) ----------------
// 0                      : float scales[2]   {x_scale, w_scale}
// 256                    : float pMax[RED_BLOCKS]
// 256 + 4*RED_BLOCKS     : float pSum[RED_BLOCKS]
// 16384                  : int8  Xq[M*K]   (fragment-major)
// 16384 + M*K            : int8  Wq[N*K]   (fragment-major)

// ---------------- reduction kernels ----------------
__global__ __launch_bounds__(256) void bl_reduce_absmax(
    const float* __restrict__ x, long long n, float* __restrict__ partial) {
  __shared__ float sm[256];
  float m = 0.0f;
  for (long long i = (long long)blockIdx.x * 256 + threadIdx.x; i < n;
       i += (long long)gridDim.x * 256)
    m = fmaxf(m, fabsf(x[i]));
  sm[threadIdx.x] = m;
  __syncthreads();
  for (int s = 128; s > 0; s >>= 1) {
    if (threadIdx.x < s) sm[threadIdx.x] = fmaxf(sm[threadIdx.x], sm[threadIdx.x + s]);
    __syncthreads();
  }
  if (threadIdx.x == 0) partial[blockIdx.x] = sm[0];
}

__global__ __launch_bounds__(256) void bl_reduce_abssum(
    const float* __restrict__ w, long long n, float* __restrict__ partial) {
  __shared__ float sm[256];
  float s = 0.0f;
  for (long long i = (long long)blockIdx.x * 256 + threadIdx.x; i < n;
       i += (long long)gridDim.x * 256)
    s += fabsf(w[i]);
  sm[threadIdx.x] = s;
  __syncthreads();
  for (int t = 128; t > 0; t >>= 1) {
    if (threadIdx.x < t) sm[threadIdx.x] += sm[threadIdx.x + t];
    __syncthreads();
  }
  if (threadIdx.x == 0) partial[blockIdx.x] = sm[0];
}

__global__ __launch_bounds__(256) void bl_finalize_scales(
    const float* __restrict__ pMax, const float* __restrict__ pSum,
    float* __restrict__ scales) {
  __shared__ float sm[256], ss[256];
  float m = 0.0f, s = 0.0f;
  for (int i = threadIdx.x; i < RED_BLOCKS; i += 256) {
    m = fmaxf(m, pMax[i]);
    s += pSum[i];
  }
  sm[threadIdx.x] = m;
  ss[threadIdx.x] = s;
  __syncthreads();
  for (int t = 128; t > 0; t >>= 1) {
    if (threadIdx.x < t) {
      sm[threadIdx.x] = fmaxf(sm[threadIdx.x], sm[threadIdx.x + t]);
      ss[threadIdx.x] += ss[threadIdx.x + t];
    }
    __syncthreads();
  }
  if (threadIdx.x == 0) {
    float absmax = fmaxf(sm[0], EPSQ);
    scales[0] = absmax / 127.0f;                                   // x_scale
    float mean = ss[0] / (float)((long long)N_DIM * K_DIM);
    scales[1] = fmaxf(mean, EPSQ);                                 // w_scale
  }
}

// ---------------- fused quantize + fragment-pack ----------------
__device__ __forceinline__ int bl_pack4(float4 v, float sc, float lo, float hi) {
  int b0 = ((int)fminf(fmaxf(rintf(v.x / sc), lo), hi)) & 255;
  int b1 = ((int)fminf(fmaxf(rintf(v.y / sc), lo), hi)) & 255;
  int b2 = ((int)fminf(fmaxf(rintf(v.z / sc), lo), hi)) & 255;
  int b3 = ((int)fminf(fmaxf(rintf(v.w / sc), lo), hi)) & 255;
  return b0 | (b1 << 8) | (b2 << 16) | (b3 << 24);
}

// One thread = one lane of one 16x64 fragment. Grid sized exactly:
// frag_count * 32 threads. Reads 32 floats (4 groups of 8), writes 32
// contiguous int8 (two b128 stores); a whole wave writes 1 KB contiguous.
__global__ __launch_bounds__(256) void bl_quant_pack(
    const float* __restrict__ src,      // [rows x K_DIM] row-major f32
    signed char* __restrict__ dst,      // fragment-major int8
    const float* __restrict__ scales, int sidx, float lo, float hi) {
  const long long gid = (long long)blockIdx.x * 256 + threadIdx.x;
  const int lane = (int)(gid & 31);
  const long long fid = gid >> 5;
  const int mt = (int)(fid / KT);
  const int kt = (int)(fid % KT);
  const int row = mt * 16 + (lane & 15);
  const int kbase = kt * 64 + (lane >> 4) * 8;
  const float sc = scales[sidx];

  const float* s0 = src + (size_t)row * K_DIM + kbase;
  int w0[8];
#pragma unroll
  for (int g = 0; g < 4; ++g) {            // K chunks at +0,+16,+32,+48
    const float4* f = (const float4*)(s0 + g * 16);
    w0[2 * g]     = bl_pack4(f[0], sc, lo, hi);
    w0[2 * g + 1] = bl_pack4(f[1], sc, lo, hi);
  }
  int4* d = (int4*)(dst + ((size_t)fid * 32 + lane) * 32);
  d[0] = make_int4(w0[0], w0[1], w0[2], w0[3]);
  d[1] = make_int4(w0[4], w0[5], w0[6], w0[7]);
}

// ---------------- WMMA int8 GEMM ----------------
__device__ __forceinline__ v8i bl_load_frag(const signed char* __restrict__ p) {
  const int4* q = (const int4*)p;       // 32-byte aligned
  int4 c0 = q[0];
  int4 c1 = q[1];
  v8i f = {c0.x, c0.y, c0.z, c0.w, c1.x, c1.y, c1.z, c1.w};
  return f;
}

// block = 256 threads = 8 waves; block tile 128(M) x 256(N)
// wave tile 64(M) x 64(N) = 4x4 accumulators of 16x16
// Per K-step of 64: 16 global_load_b128 feed 16 v_wmma_i32_16x16x64_iu8.
__global__ __launch_bounds__(256) void bl_wmma_gemm(
    const signed char* __restrict__ Aq,   // fragment-major [M/16][KT][32][32]
    const signed char* __restrict__ Wq,   // fragment-major [N/16][KT][32][32]
    const float* __restrict__ scales,
    float* __restrict__ out)              // [M_DIM x N_DIM] fp32
{
  const int tid  = threadIdx.x;
  const int lane = tid & 31;
  const int l    = lane & 15;
  const int half = lane >> 4;
  const int wid  = tid >> 5;
  const int waveM = wid & 1;   // 2 waves along M
  const int waveN = wid >> 1;  // 4 waves along N

  const int m0 = blockIdx.y * 128 + waveM * 64;
  const int n0 = blockIdx.x * 256 + waveN * 64;

  const signed char* ap = Aq + (size_t)(m0 >> 4) * MT_STRIDE + (size_t)lane * 32;
  const signed char* bp = Wq + (size_t)(n0 >> 4) * MT_STRIDE + (size_t)lane * 32;

  const v8i vz = {0, 0, 0, 0, 0, 0, 0, 0};
  v8i acc[4][4];
#pragma unroll
  for (int mi = 0; mi < 4; ++mi)
#pragma unroll
    for (int t = 0; t < 4; ++t) acc[mi][t] = vz;

  for (int kt = 0; kt < KT; ++kt) {
    const size_t ko = (size_t)kt * FRAG_BYTES;
    v8i a0 = bl_load_frag(ap + ko);
    v8i a1 = bl_load_frag(ap + MT_STRIDE + ko);
    v8i a2 = bl_load_frag(ap + 2 * MT_STRIDE + ko);
    v8i a3 = bl_load_frag(ap + 3 * MT_STRIDE + ko);
#pragma unroll
    for (int t = 0; t < 4; ++t) {
      v8i b = bl_load_frag(bp + (size_t)t * MT_STRIDE + ko);
      acc[0][t] = __builtin_amdgcn_wmma_i32_16x16x64_iu8(
          true, a0, true, b, acc[0][t], false, false);
      acc[1][t] = __builtin_amdgcn_wmma_i32_16x16x64_iu8(
          true, a1, true, b, acc[1][t], false, false);
      acc[2][t] = __builtin_amdgcn_wmma_i32_16x16x64_iu8(
          true, a2, true, b, acc[2][t], false, false);
      acc[3][t] = __builtin_amdgcn_wmma_i32_16x16x64_iu8(
          true, a3, true, b, acc[3][t], false, false);
    }
  }

  // Epilogue: scale to fp32 and store nontemporally (output is write-once,
  // 361 MB; NT stores keep Xq/Wq resident in the 192 MB L2 across blocks).
  const float s = scales[0] * scales[1];
#pragma unroll
  for (int mi = 0; mi < 4; ++mi) {
    const int rowBase = m0 + mi * 16 + half * 8;   // VGPR r -> row rowBase + r
#pragma unroll
    for (int t = 0; t < 4; ++t) {
      float* o = out + (size_t)rowBase * N_DIM + (n0 + t * 16 + l);
#pragma unroll
      for (int r = 0; r < 8; ++r)
        __builtin_nontemporal_store((float)acc[mi][t][r] * s,
                                    o + (size_t)r * N_DIM);
    }
  }
}

// ---------------- host-side launch ----------------
extern "C" void kernel_launch(void* const* d_in, const int* in_sizes, int n_in,
                              void* d_out, int out_size, void* d_ws, size_t ws_size,
                              hipStream_t stream) {
  (void)in_sizes; (void)n_in; (void)out_size; (void)ws_size;

  const float* x = (const float*)d_in[0];   // (4,2048,4096) f32
  const float* w = (const float*)d_in[1];   // (11008,4096) f32
  float* out = (float*)d_out;               // (4,2048,11008) f32

  char* ws = (char*)d_ws;
  float* scales = (float*)ws;
  float* pMax   = (float*)(ws + 256);
  float* pSum   = (float*)(ws + 256 + RED_BLOCKS * 4);
  signed char* Xq = (signed char*)(ws + 16384);
  signed char* Wq = (signed char*)(ws + 16384 + (size_t)M_DIM * K_DIM);

  const long long nx = (long long)M_DIM * K_DIM;
  const long long nw = (long long)N_DIM * K_DIM;

  bl_reduce_absmax<<<RED_BLOCKS, 256, 0, stream>>>(x, nx, pMax);
  bl_reduce_abssum<<<RED_BLOCKS, 256, 0, stream>>>(w, nw, pSum);
  bl_finalize_scales<<<1, 256, 0, stream>>>(pMax, pSum, scales);

  // X: (8192/16)*(4096/64) = 32768 frags * 32 lanes = 1,048,576 threads
  bl_quant_pack<<<4096, 256, 0, stream>>>(x, Xq, scales, 0, -128.0f, 127.0f);
  // W: (11008/16)*(4096/64) = 44032 frags * 32 lanes = 1,409,024 threads
  bl_quant_pack<<<5504, 256, 0, stream>>>(w, Wq, scales, 1, -1.0f, 1.0f);

  dim3 grid(N_DIM / 256, M_DIM / 128);   // (43, 64)
  bl_wmma_gemm<<<grid, 256, 0, stream>>>(Xq, Wq, scales, out);
}
